// HandlePredictorSWTpl_85066122265629
// MI455X (gfx1250) — matrix operand order repaired and checked by reading, not compiled, not run on hardware
//
#include <hip/hip_runtime.h>
#include <hip/hip_bf16.h>
#include <math.h>

// ---------------- problem constants ----------------
#define NNODES 50000
#define NEDGES 300000
#define NGRAPH 10

// ---------------- GEMM tile config -----------------
#define BM 128
#define BN 64
#define KC 64
#define ASTR 66   // A LDS stride (bf16) - bank-conflict padding
#define BSTR 72   // B LDS stride (bf16) - 144B row: 16B-aligned for async B128

typedef __attribute__((ext_vector_type(16))) __bf16 v16bf;
typedef __attribute__((ext_vector_type(8)))  float  v8f;

union AFrag { v16bf v; unsigned u[8]; };
union CFrag { v8f   v; float    f[8]; };

__device__ __forceinline__ float bf2f(unsigned short h) {
  return __uint_as_float(((unsigned)h) << 16);
}
__device__ __forceinline__ unsigned short f2bf(float f) {
  unsigned u = __float_as_uint(f);
  unsigned r = ((u >> 16) & 1u) + 0x7FFFu; // round-to-nearest-even
  return (unsigned short)((u + r) >> 16);
}
// monotonic float -> uint encoding (for atomic max on floats)
__device__ __forceinline__ unsigned encf(float f) {
  unsigned u = __float_as_uint(f);
  return (u & 0x80000000u) ? ~u : (u | 0x80000000u);
}

// ---------------- tiny utility kernels ----------------
__global__ void k_zero(float* p, int n) {
  int i = blockIdx.x * blockDim.x + threadIdx.x;
  if (i < n) p[i] = 0.0f;
}
__global__ void k_fill_u32(unsigned* p, unsigned v, int n) {
  int i = blockIdx.x * blockDim.x + threadIdx.x;
  if (i < n) p[i] = v;
}

// Fold previous-layer BN affine (aF,cF over K) into weights.
// Emits TRANSPOSED zero-padded bf16 weights Wt[Np][Kp] (Kp,Np multiples of 64)
// so the GEMM can stage B tiles with contiguous async global->LDS B128 copies.
__global__ void k_fold(const float* __restrict__ W, const float* __restrict__ b,
                       const float* __restrict__ aF, const float* __restrict__ cF,
                       int K, int N, int Kp,
                       unsigned short* __restrict__ Wt, float* __restrict__ bE) {
  int j = blockIdx.x * blockDim.x + threadIdx.x; // runs over Np
  float acc = (j < N) ? b[j] : 0.0f;
  for (int k = 0; k < Kp; ++k) {
    unsigned short v = 0;
    if (j < N && k < K) {
      float w  = W[(size_t)k * N + j];
      float av = aF ? aF[k] : 1.0f;
      float cv = cF ? cF[k] : 0.0f;
      acc += cv * w;
      v = f2bf(av * w);
    }
    Wt[(size_t)j * Kp + k] = v;
  }
  if (j < N) bE[j] = acc;
}

// stats[0..255]=sum, stats[256..511]=sumsq  ->  a = gamma*rsqrt(var+eps), c = beta - mu*a
__global__ void k_finalize(const float* __restrict__ stats, float invM,
                           const float* __restrict__ gamma, const float* __restrict__ beta,
                           float* __restrict__ aO, float* __restrict__ cO, int N) {
  int j = blockIdx.x * blockDim.x + threadIdx.x;
  if (j >= N) return;
  float mu  = stats[j] * invM;
  float var = stats[256 + j] * invM - mu * mu;
  float rs  = rsqrtf(var + 1e-5f);
  float a   = gamma[j] * rs;
  aO[j] = a;
  cO[j] = beta[j] - mu * a;
}

// a/c for edge-conv concat input: scale = [a,a], shift = [c,0]
__global__ void k_dup_ac(const float* a, const float* c, int F, float* aF, float* cF) {
  int k = blockIdx.x * blockDim.x + threadIdx.x;
  if (k >= 2 * F) return;
  aF[k] = a[(k < F) ? k : (k - F)];
  cF[k] = (k < F) ? c[k] : 0.0f;
}
// a/c for [x1(64), x2(128), x3(256)] concat
__global__ void k_concat3_ac(const float* a1, const float* c1, const float* a2,
                             const float* c2, const float* a3, const float* c3,
                             float* aF, float* cF) {
  int k = blockIdx.x * blockDim.x + threadIdx.x;
  if (k >= 448) return;
  if (k < 64)       { aF[k] = a1[k];       cF[k] = c1[k]; }
  else if (k < 192) { aF[k] = a2[k - 64];  cF[k] = c2[k - 64]; }
  else              { aF[k] = a3[k - 192]; cF[k] = c3[k - 192]; }
}

// apply BN affine elementwise, then segment-max scatter (encoded atomic max)
__global__ void k_scatter_max(const unsigned short* __restrict__ h,
                              const int* __restrict__ edst,
                              const float* __restrict__ a, const float* __restrict__ c,
                              int F, int lgF, unsigned* __restrict__ agg, int total) {
  int g = blockIdx.x * blockDim.x + threadIdx.x;
  if (g >= total) return;
  int e = g >> lgF, col = g & (F - 1);
  float v = a[col] * bf2f(h[(size_t)e * F + col]) + c[col];
  atomicMax(&agg[(size_t)edst[e] * F + col], encf(v));
}

// decode + where(isfinite, out, 0) -> bf16
__global__ void k_fixup_agg(const unsigned* __restrict__ agg,
                            unsigned short* __restrict__ out, int n) {
  int i = blockIdx.x * blockDim.x + threadIdx.x;
  if (i >= n) return;
  unsigned e = agg[i];
  int iv = (e & 0x80000000u) ? (int)(e & 0x7FFFFFFFu) : (int)(~e);
  float f = __int_as_float(iv);
  if (!isfinite(f)) f = 0.0f;
  out[i] = f2bf(f);
}

// ---------------- WMMA GEMM: out = relu(A @ W + b), bf16 store + BN stats ----------------
// mode 0: A0 bf16 direct (row stride F == K)
// mode 1: edge gather from bf16 node feats (F = feat dim, K = 2F)
// mode 2: edge gather from f32 node feats
// mode 3: concat3 of x1(64)/x2(128)/x3(256)
__device__ __forceinline__ float fetchA(int mode, int row, int k,
    const unsigned short* A0, const unsigned short* A1, const unsigned short* A2,
    const float* Af, const int* es, const int* ed, int F) {
  if (mode == 0) return bf2f(A0[(size_t)row * F + k]);
  if (mode == 1) {
    int d = ed[row];
    if (k < F) return bf2f(A0[(size_t)d * F + k]);
    int s = es[row], kk = k - F;
    return bf2f(A0[(size_t)s * F + kk]) - bf2f(A0[(size_t)d * F + kk]);
  }
  if (mode == 2) {
    int d = ed[row];
    if (k < F) return Af[(size_t)d * F + k];
    int s = es[row], kk = k - F;
    return Af[(size_t)s * F + kk] - Af[(size_t)d * F + kk];
  }
  if (k < 64)  return bf2f(A0[(size_t)row * 64 + k]);
  if (k < 192) return bf2f(A1[(size_t)row * 128 + (k - 64)]);
  return bf2f(A2[(size_t)row * 256 + (k - 192)]);
}

__global__ __launch_bounds__(256) void k_gemm(
    int mode,
    const unsigned short* __restrict__ A0, const unsigned short* __restrict__ A1,
    const unsigned short* __restrict__ A2, const float* __restrict__ Af,
    const int* __restrict__ esrc, const int* __restrict__ edst,
    int M, int K, int N, int F, int Kp,
    const unsigned short* __restrict__ Wt, const float* __restrict__ bE,
    unsigned short* __restrict__ outB, float* __restrict__ stats) {
  __shared__ __align__(16) unsigned short As[BM * ASTR];
  __shared__ __align__(16) unsigned short Bs[BN * BSTR];

  const int tid = threadIdx.x, lane = tid & 31, wid = tid >> 5;
  const int m0 = blockIdx.x * BM, n0 = blockIdx.y * BN;

  v8f acc[4];
  { CFrag z;
    #pragma unroll
    for (int q = 0; q < 8; ++q) z.f[q] = 0.0f;
    #pragma unroll
    for (int t = 0; t < 4; ++t) acc[t] = z.v; }

  for (int k0 = 0; k0 < K; k0 += KC) {
    __syncthreads();
    // stage A tile (BM x KC) via compute path (gather / concat / direct), zero-padded
    for (int t = tid; t < BM * KC; t += 256) {
      int r = t >> 6, kk = t & 63;
      int row = m0 + r, k = k0 + kk;
      float v = 0.0f;
      if (row < M && k < K) v = fetchA(mode, row, k, A0, A1, A2, Af, esrc, edst, F);
      As[r * ASTR + kk] = f2bf(v);
    }
    // stage B tile with async global->LDS DMA: Wt is transposed + padded, so each
    // lane copies 16 contiguous bytes (8 bf16 along K) with no bounds checks.
    // 64 rows x 8 xfers = 512 transfers; 256 threads -> 2 each. Tracked by ASYNCcnt.
    for (int t = tid; t < BN * (KC / 8); t += 256) {
      int n = t >> 3, q = t & 7;
      unsigned lds = (unsigned)(size_t)&Bs[n * BSTR + q * 8];
      unsigned long long g =
          (unsigned long long)(size_t)(Wt + (size_t)(n0 + n) * Kp + (k0 + q * 8));
      asm volatile("global_load_async_to_lds_b128 %0, %1, off"
                   :: "v"(lds), "v"(g) : "memory");
    }
    asm volatile("s_wait_asynccnt 0x0" ::: "memory");
    __syncthreads();

    const int rowb = lane & 15;
    for (int ks = 0; ks < KC; ks += 32) {
      if (k0 + ks >= K) break;
      // A fragment: 16x32 bf16 (ISA 16-bit A layout)
      AFrag af;
      #pragma unroll
      for (int v = 0; v < 8; ++v) {
        int kk = ks + ((v < 4) ? 0 : 16) + ((lane < 16) ? 0 : 8) + 2 * (v & 3);
        af.u[v] = *(const unsigned*)&As[(wid * 16 + rowb) * ASTR + kk];
      }
      #pragma unroll
      for (int t = 0; t < 4; ++t) {
        // B fragment: 32x16 bf16 (ISA 16-bit B layout), column = t*16 + lane%16
        AFrag bfr;
        #pragma unroll
        for (int v = 0; v < 8; ++v) {
          int kk = ks + ((v < 4) ? 0 : 16) + ((lane < 16) ? 0 : 8) + 2 * (v & 3);
          bfr.u[v] = *(const unsigned*)&Bs[(t * 16 + rowb) * BSTR + kk];
        }
        acc[t] = __builtin_amdgcn_wmma_f32_16x16x32_bf16(
            false, af.v, false, bfr.v, (short)0, acc[t], false, false);
      }
    }
  }

  // epilogue: bias + relu, bf16 store, per-column sum/sumsq -> atomics
  const int rowbase = m0 + wid * 16;
  #pragma unroll
  for (int t = 0; t < 4; ++t) {
    int col = n0 + t * 16 + (lane & 15);
    CFrag cu; cu.v = acc[t];
    float bias = (col < N) ? bE[col] : 0.0f;
    float s = 0.0f, s2 = 0.0f;
    #pragma unroll
    for (int i = 0; i < 8; ++i) {
      int row = rowbase + ((lane < 16) ? i : (8 + i));
      float val = fmaxf(cu.f[i] + bias, 0.0f);
      if (row < M && col < N) {
        outB[(size_t)row * N + col] = f2bf(val);
        s += val; s2 += val * val;
      }
    }
    s  += __shfl_xor(s, 16, 32);   // combine rows 0-7 with 8-15 (same column)
    s2 += __shfl_xor(s2, 16, 32);
    if (stats && lane < 16 && col < N) {
      atomicAdd(&stats[col], s);
      atomicAdd(&stats[256 + col], s2);
    }
  }
}

// ---------------- head: 128->24 + softmax + segment sum ----------------
// Wt is transposed padded [64][128] (only first 24 rows valid)
__global__ void k_mlp3(const unsigned short* __restrict__ h,
                       const unsigned short* __restrict__ Wt, const float* __restrict__ bE,
                       const int* __restrict__ batch,
                       float* __restrict__ logits, float* __restrict__ sw,
                       float* __restrict__ seg) {
  int i = blockIdx.x * blockDim.x + threadIdx.x;
  if (i >= NNODES) return;
  float acc[24];
  #pragma unroll
  for (int j = 0; j < 24; ++j) acc[j] = bE[j];
  const unsigned short* hr = h + (size_t)i * 128;
  for (int k = 0; k < 128; ++k) {
    float hv = bf2f(hr[k]);
    #pragma unroll
    for (int j = 0; j < 24; ++j) acc[j] += hv * bf2f(Wt[(size_t)j * 128 + k]);
  }
  float m = -1e30f;
  #pragma unroll
  for (int j = 0; j < 24; ++j) m = fmaxf(m, acc[j]);
  float ex[24], s = 0.0f;
  #pragma unroll
  for (int j = 0; j < 24; ++j) { ex[j] = expf(acc[j] - m); s += ex[j]; }
  float inv = 1.0f / s;
  int b = batch[i];
  #pragma unroll
  for (int j = 0; j < 24; ++j) {
    logits[(size_t)i * 24 + j] = acc[j];
    float v = ex[j] * inv;
    sw[(size_t)i * 24 + j] = v;
    atomicAdd(&seg[b * 24 + j], v);
  }
}

__global__ void k_score(const float* __restrict__ sw, const float* __restrict__ seg,
                        const int* __restrict__ batch, const float* __restrict__ x,
                        float* __restrict__ score, float* __restrict__ wpos) {
  int i = blockIdx.x * blockDim.x + threadIdx.x;
  if (i >= NNODES) return;
  int b = batch[i];
  float p0 = x[(size_t)i * 6 + 0], p1 = x[(size_t)i * 6 + 1], p2 = x[(size_t)i * 6 + 2];
  #pragma unroll
  for (int j = 0; j < 24; ++j) {
    float sc = sw[(size_t)i * 24 + j] / seg[b * 24 + j];
    score[(size_t)i * 24 + j] = sc;
    atomicAdd(&wpos[b * 72 + j * 3 + 0], sc * p0);
    atomicAdd(&wpos[b * 72 + j * 3 + 1], sc * p1);
    atomicAdd(&wpos[b * 72 + j * 3 + 2], sc * p2);
  }
}

// ---------------- host orchestration ----------------
struct Lin { const float *W, *b, *g, *be; int K, N; };

extern "C" void kernel_launch(void* const* d_in, const int* in_sizes, int n_in,
                              void* d_out, int out_size, void* d_ws, size_t ws_size,
                              hipStream_t stream) {
  (void)in_sizes; (void)n_in; (void)out_size; (void)ws_size;
  const float* x    = (const float*)d_in[0];
  const int*   eidx = (const int*)d_in[1];
  const int*   esrc = eidx;
  const int*   edst = eidx + NEDGES;
  const int*   batch = (const int*)d_in[2];
  auto P = [&](int i) { return (const float*)d_in[i]; };

  // params in setup_inputs insertion order: per lin_bn -> W, b, gamma, beta
  Lin g1e0{P(3),  P(4),  P(5),  P(6),  12, 32};
  Lin g1e1{P(7),  P(8),  P(9),  P(10), 32, 32};
  Lin g1n {P(11), P(12), P(13), P(14), 32, 64};
  Lin g2e0{P(15), P(16), P(17), P(18), 128, 64};
  Lin g2e1{P(19), P(20), P(21), P(22), 64, 64};
  Lin g2n {P(23), P(24), P(25), P(26), 64, 128};
  Lin g3e0{P(27), P(28), P(29), P(30), 256, 128};
  Lin g3e1{P(31), P(32), P(33), P(34), 128, 128};
  Lin g3n {P(35), P(36), P(37), P(38), 128, 256};
  Lin glbL{P(39), P(40), P(41), P(42), 448, 256};
  Lin m20 {P(43), P(44), P(45), P(46), 256, 256};
  Lin m21 {P(47), P(48), P(49), P(50), 256, 128};
  const float* W3 = P(51);
  const float* b3 = P(52);

  // workspace bump allocator
  char* wp = (char*)d_ws;
  auto alloc = [&](size_t bytes) { void* r = wp; wp += (bytes + 255) & ~(size_t)255; return r; };
  float* stats  = (float*)alloc(512 * 4);
  float* aFull  = (float*)alloc(448 * 4);
  float* cFull  = (float*)alloc(448 * 4);
  float* bEff   = (float*)alloc(256 * 4);
  float* aT1 = (float*)alloc(256*4); float* cT1 = (float*)alloc(256*4);
  float* aT2 = (float*)alloc(256*4); float* cT2 = (float*)alloc(256*4);
  float* a1 = (float*)alloc(256*4); float* c1 = (float*)alloc(256*4);
  float* a2 = (float*)alloc(256*4); float* c2 = (float*)alloc(256*4);
  float* a3 = (float*)alloc(256*4); float* c3 = (float*)alloc(256*4);
  float* a4 = (float*)alloc(256*4); float* c4 = (float*)alloc(256*4);
  float* a5 = (float*)alloc(256*4); float* c5 = (float*)alloc(256*4);
  float* a6 = (float*)alloc(256*4); float* c6 = (float*)alloc(256*4);
  float* seg = (float*)alloc(NGRAPH * 24 * 4);
  unsigned short* Wbf    = (unsigned short*)alloc((size_t)448 * 256 * 2); // transposed padded
  unsigned short* hA     = (unsigned short*)alloc((size_t)NEDGES * 128 * 2);
  unsigned short* hB     = (unsigned short*)alloc((size_t)NEDGES * 128 * 2);
  unsigned*       aggEnc = (unsigned*)alloc((size_t)NNODES * 128 * 4);
  unsigned short* aggBf  = (unsigned short*)alloc((size_t)NNODES * 128 * 2);
  unsigned short* x1b = (unsigned short*)alloc((size_t)NNODES * 64 * 2);
  unsigned short* x2b = (unsigned short*)alloc((size_t)NNODES * 128 * 2);
  unsigned short* x3b = (unsigned short*)alloc((size_t)NNODES * 256 * 2);
  unsigned short* x4b = (unsigned short*)alloc((size_t)NNODES * 256 * 2);
  unsigned short* h5b = (unsigned short*)alloc((size_t)NNODES * 256 * 2);
  unsigned short* h6b = (unsigned short*)alloc((size_t)NNODES * 128 * 2);

  float* out    = (float*)d_out;
  float* scoreO = out;
  float* wposO  = out + (size_t)NNODES * 24;
  float* logitO = wposO + 720;
  float* swO    = logitO + (size_t)NNODES * 24;

  auto run_layer = [&](int mode, const unsigned short* A0, const unsigned short* Aa,
                       const unsigned short* Ab, const float* Af, int M, int F,
                       const Lin& L, const float* aF, const float* cF,
                       unsigned short* outB, float* aO, float* cO) {
    int Kp = (L.K + 63) & ~63;
    int Np = (L.N + 63) & ~63;
    k_zero<<<2, 256, 0, stream>>>(stats, 512);
    k_fold<<<Np / 64, 64, 0, stream>>>(L.W, L.b, aF, cF, L.K, L.N, Kp, Wbf, bEff);
    dim3 grid((M + BM - 1) / BM, (L.N + BN - 1) / BN);
    k_gemm<<<grid, 256, 0, stream>>>(mode, A0, Aa, Ab, Af, esrc, edst,
                                     M, L.K, L.N, F, Kp, Wbf, bEff, outB, stats);
    k_finalize<<<(L.N + 63) / 64, 64, 0, stream>>>(stats, 1.0f / M, L.g, L.be, aO, cO, L.N);
  };

  auto edge_conv = [&](const unsigned short* xb, const float* xf, int F,
                       const float* aPrev, const float* cPrev,
                       const Lin& L0, const Lin& L1) {
    if (xf) {
      run_layer(2, nullptr, nullptr, nullptr, xf, NEDGES, F, L0, nullptr, nullptr, hA, aT1, cT1);
    } else {
      k_dup_ac<<<(2 * F + 63) / 64, 64, 0, stream>>>(aPrev, cPrev, F, aFull, cFull);
      run_layer(1, xb, nullptr, nullptr, nullptr, NEDGES, F, L0, aFull, cFull, hA, aT1, cT1);
    }
    run_layer(0, hA, nullptr, nullptr, nullptr, NEDGES, L0.N, L1, aT1, cT1, hB, aT2, cT2);
    int F2 = L1.N;
    int lg = (F2 == 32) ? 5 : (F2 == 64) ? 6 : 7;
    int ntot = NNODES * F2, etot = NEDGES * F2;
    k_fill_u32<<<(ntot + 255) / 256, 256, 0, stream>>>(aggEnc, 0x007FFFFFu, ntot); // enc(-inf)
    k_scatter_max<<<(etot + 255) / 256, 256, 0, stream>>>(hB, edst, aT2, cT2, F2, lg, aggEnc, etot);
    k_fixup_agg<<<(ntot + 255) / 256, 256, 0, stream>>>(aggEnc, aggBf, ntot);
  };

  // gcu1 / gcu2 / gcu3
  edge_conv(nullptr, x, 6, nullptr, nullptr, g1e0, g1e1);
  run_layer(0, aggBf, nullptr, nullptr, nullptr, NNODES, 32, g1n, nullptr, nullptr, x1b, a1, c1);
  edge_conv(x1b, nullptr, 64, a1, c1, g2e0, g2e1);
  run_layer(0, aggBf, nullptr, nullptr, nullptr, NNODES, 64, g2n, nullptr, nullptr, x2b, a2, c2);
  edge_conv(x2b, nullptr, 128, a2, c2, g3e0, g3e1);
  run_layer(0, aggBf, nullptr, nullptr, nullptr, NNODES, 128, g3n, nullptr, nullptr, x3b, a3, c3);

  // glb on concat(x1,x2,x3), then mlp2
  k_concat3_ac<<<7, 64, 0, stream>>>(a1, c1, a2, c2, a3, c3, aFull, cFull);
  run_layer(3, x1b, x2b, x3b, nullptr, NNODES, 0, glbL, aFull, cFull, x4b, a4, c4);
  run_layer(0, x4b, nullptr, nullptr, nullptr, NNODES, 256, m20, a4, c4, h5b, a5, c5);
  run_layer(0, h5b, nullptr, nullptr, nullptr, NNODES, 256, m21, a5, c5, h6b, a6, c6);

  // head: fold h6 BN into mlp3 weights (transposed padded 64x128), softmax, segment sums
  k_fold<<<1, 64, 0, stream>>>(W3, b3, a6, c6, 128, 24, 128, Wbf, bEff);
  k_zero<<<1, 256, 0, stream>>>(seg, NGRAPH * 24);
  k_zero<<<3, 256, 0, stream>>>(wposO, 720);
  k_mlp3<<<(NNODES + 127) / 128, 128, 0, stream>>>(h6b, Wbf, bEff, batch, logitO, swO, seg);
  k_score<<<(NNODES + 127) / 128, 128, 0, stream>>>(swO, seg, batch, x, scoreO, wposO);
}
